// RNNCFModel_22539988369864
// MI455X (gfx1250) — compile-verified, matrix operand order checked by reading, not run on hardware
//
#include <hip/hip_runtime.h>

typedef __attribute__((ext_vector_type(16))) _Float16 v16h;
typedef __attribute__((ext_vector_type(8)))  _Float16 v8h;
typedef __attribute__((ext_vector_type(8)))  float    v8f;

#define PASTN 25
#define NTIME 256
#define NVEH  4096
#define WAVES_PER_BLOCK 4

// Model constants
#define MAXHD_INV (1.0f / 200.0f)
#define MAXV_INV  (1.0f / 40.0f)
#define DTC       0.1f

__global__ __launch_bounds__(128) void rnncf_kernel(
    const float* __restrict__ lead,   // (NVEH, 280, 2)
    const float* __restrict__ cur,    // (NVEH, 25, 2)
    const float* __restrict__ w1,     // (3,3,32)
    const float* __restrict__ b1,     // (32)
    const float* __restrict__ w2,     // (3,32,64)
    const float* __restrict__ b2,     // (64)
    const float* __restrict__ d2w,    // (384,10)
    const float* __restrict__ d2b,    // (10)
    const float* __restrict__ d1w,    // (10)
    const float* __restrict__ d1b,    // (1)
    float* __restrict__ out)          // (NVEH, 256, 2)
{
    // ---- shared weights (block-wide) ----
    __shared__ float s_w1[3][3][32];
    __shared__ float s_b1[32];
    __shared__ __align__(16) _Float16 s_w2t[64][96];   // [N][K], K = tap*32 + cin
    __shared__ float s_b2[64];
    __shared__ float s_d2w[384][10];
    __shared__ float s_d2b[10];
    __shared__ float s_d1w[10];
    __shared__ float s_d1b;

    // ---- per-wave working state ----
    __shared__ __align__(16) float s_lead[WAVES_PER_BLOCK][280][2];
    __shared__ float s_cpos[WAVES_PER_BLOCK][32];      // ring buffer (25 live)
    __shared__ float s_cspd[WAVES_PER_BLOCK][32];
    __shared__ __align__(16) _Float16 s_h[WAVES_PER_BLOCK][18][32]; // pooled conv1 out (13 live rows, rest zero)
    __shared__ float s_x[WAVES_PER_BLOCK][28][3];      // conv1 input, rows 1..25 live, 0/26/27 zero pad
    __shared__ float s_c2[WAVES_PER_BLOCK][16][64];    // conv2+relu output (rows 0..10 live, 11..15 pad)

    const int tid  = threadIdx.x;
    const int wave = tid >> 5;
    const int lane = tid & 31;
    const int v    = blockIdx.x * WAVES_PER_BLOCK + wave;

    // ---- one-time setup: weights into LDS (f16 transpose for WMMA B operand) ----
    for (int i = tid; i < 3 * 3 * 32; i += 128) (&s_w1[0][0][0])[i] = w1[i];
    for (int i = tid; i < 32; i += 128) s_b1[i] = b1[i];
    for (int i = tid; i < 3 * 32 * 64; i += 128) {
        int k = i / (32 * 64), r = i % (32 * 64);
        int ci = r / 64, n = r % 64;
        s_w2t[n][k * 32 + ci] = (_Float16)w2[i];
    }
    for (int i = tid; i < 64; i += 128) s_b2[i] = b2[i];
    for (int i = tid; i < 384 * 10; i += 128) (&s_d2w[0][0])[i] = d2w[i];
    if (tid < 10) { s_d2b[tid] = d2b[tid]; s_d1w[tid] = d1w[tid]; }
    if (tid == 0) s_d1b = d1b[0];

    // ---- per-wave: lead series into LDS via CDNA5 async global->LDS DMA ----
    // 280*2 floats = 2240 B = 140 x b128 chunks, lane-strided, ASYNCcnt-tracked.
    {
        unsigned long long gbase =
            (unsigned long long)(const void*)(lead + (size_t)v * 560);
        unsigned lbase = (unsigned)(unsigned long long)(uintptr_t)&s_lead[wave][0][0];
        for (int chunk = lane; chunk < 140; chunk += 32) {
            unsigned long long gaddr = gbase + (unsigned long long)chunk * 16ull;
            unsigned laddr = lbase + (unsigned)chunk * 16u;
            asm volatile("global_load_async_to_lds_b128 %0, %1, off"
                         :
                         : "v"(laddr), "v"(gaddr)
                         : "memory");
        }
        asm volatile("s_wait_asynccnt 0x0" ::: "memory");
    }

    if (lane < PASTN) {
        s_cpos[wave][lane] = cur[(v * PASTN + lane) * 2 + 0];
        s_cspd[wave][lane] = cur[(v * PASTN + lane) * 2 + 1];
    }
    // zero A-matrix padding rows (13..17) of pooled activations
    for (int i = lane; i < 5 * 32; i += 32)
        s_h[wave][13 + (i >> 5)][i & 31] = (_Float16)0.0f;
    // zero conv1 input padding rows
    if (lane < 3) {
        s_x[wave][0][lane]  = 0.0f;
        s_x[wave][26][lane] = 0.0f;
        s_x[wave][27][lane] = 0.0f;
    }
    __syncthreads();

    const int arow = lane & 15;            // WMMA fragment row (M for A, N-in-tile for B)
    const int colb = (lane >> 4) << 3;     // 0 or 8: K sub-block select per ISA f16 layout
    const int mbase = (lane >> 4) << 3;    // C layout: VGPR j -> M = j + mbase

    int head = 0;
    for (int t = 0; t < NTIME; ++t) {
        // -------- build normalized CNN input window (25,3) --------
        if (lane < PASTN) {
            float lp = s_lead[wave][t + lane][0];
            float ls = s_lead[wave][t + lane][1];
            float cp = s_cpos[wave][(head + lane) & 31];
            float cs = s_cspd[wave][(head + lane) & 31];
            s_x[wave][lane + 1][0] = (lp - cp) * MAXHD_INV;
            s_x[wave][lane + 1][1] = cs * MAXV_INV;
            s_x[wave][lane + 1][2] = ls * MAXV_INV;
        }
        __syncthreads();

        // -------- conv1(SAME,K3,3->32) + ReLU + maxpool2 -> h[13][32] (f16) --------
        // K=9 too small for WMMA (72% pad waste) -> VALU, co-executes with XDL.
        for (int idx = lane; idx < 13 * 32; idx += 32) {
            int i  = idx >> 5;
            int ch = idx & 31;
            int p0 = 2 * i;
            float o0 = s_b1[ch];
            float o1 = o0;
            #pragma unroll
            for (int k = 0; k < 3; ++k) {
                #pragma unroll
                for (int ci = 0; ci < 3; ++ci) {
                    float w = s_w1[k][ci][ch];
                    o0 = fmaf(s_x[wave][p0 + k][ci], w, o0);
                    o1 = fmaf(s_x[wave][p0 + 1 + k][ci], w, o1);
                }
            }
            o0 = fmaxf(o0, 0.0f);
            float m = (p0 + 1 < PASTN) ? fmaxf(o0, fmaxf(o1, 0.0f)) : o0;
            s_h[wave][i][ch] = (_Float16)m;
        }
        __syncthreads();

        // -------- conv2(VALID,K3,32->64) as GEMM: M=16(pad of 11), K=96, N=64 --------
        // A fragments: row M=arow, halves 0..7 = K[colb..colb+7], 8..15 = K[colb+16..colb+23]
        v16h afrag[3];
        #pragma unroll
        for (int kt = 0; kt < 3; ++kt) {
            v8h lo = *(const v8h*)&s_h[wave][arow + kt][colb];
            v8h hi = *(const v8h*)&s_h[wave][arow + kt][colb + 16];
            afrag[kt] = __builtin_shufflevector(lo, hi,
                0, 1, 2, 3, 4, 5, 6, 7, 8, 9, 10, 11, 12, 13, 14, 15);
        }

        #pragma unroll
        for (int ntile = 0; ntile < 4; ++ntile) {
            v8f c = {};
            #pragma unroll
            for (int kt = 0; kt < 3; ++kt) {
                const _Float16* brow = &s_w2t[ntile * 16 + arow][kt * 32 + colb];
                v8h lo = *(const v8h*)brow;
                v8h hi = *(const v8h*)(brow + 16);
                v16h bfrag = __builtin_shufflevector(lo, hi,
                    0, 1, 2, 3, 4, 5, 6, 7, 8, 9, 10, 11, 12, 13, 14, 15);
                c = __builtin_amdgcn_wmma_f32_16x16x32_f16(
                        false, afrag[kt], false, bfrag, (short)0, c, false, false);
            }
            // C layout: VGPR j -> M = j + mbase; N = ntile*16 + (lane&15).
            // Rows are 256B apart -> straight-line ds_store_2addr_stride64 pairs,
            // no exec-mask divergence (rows 11..15 are dead padding).
            int n = ntile * 16 + arow;
            float bias = s_b2[n];
            #pragma unroll
            for (int j = 0; j < 8; ++j) {
                s_c2[wave][j + mbase][n] = fmaxf(c[j] + bias, 0.0f);
            }
        }
        __syncthreads();

        // -------- maxpool2 -> flatten(384) -> dense2(384x10)+ReLU -> dense1(10x1) --------
        float acc[10];
        #pragma unroll
        for (int o = 0; o < 10; ++o) acc[o] = 0.0f;
        for (int f = lane; f < 384; f += 32) {
            int i  = f >> 6;
            int ch = f & 63;
            float a0 = s_c2[wave][2 * i][ch];
            float val = (2 * i + 1 < 11) ? fmaxf(a0, s_c2[wave][2 * i + 1][ch]) : a0;
            #pragma unroll
            for (int o = 0; o < 10; ++o) acc[o] = fmaf(val, s_d2w[f][o], acc[o]);
        }
        #pragma unroll
        for (int off = 16; off > 0; off >>= 1) {
            #pragma unroll
            for (int o = 0; o < 10; ++o) acc[o] += __shfl_xor(acc[o], off, 32);
        }
        float a1 = s_d1b;
        #pragma unroll
        for (int o = 0; o < 10; ++o) {
            float z = fmaxf(acc[o] + s_d2b[o], 0.0f);
            a1 = fmaf(z, s_d1w[o], a1);
        }
        float accel = 10.0f * a1 - 6.0f;  // (MAXA-MINA)*x + MINA

        // -------- state update + output --------
        float lastp = s_cpos[wave][(head + 24) & 31];
        float lasts = s_cspd[wave][(head + 24) & 31];
        float pp = lastp + DTC * lasts;
        float ps = lasts + DTC * accel;
        if (lane == 0) {
            out[(v * NTIME + t) * 2 + 0] = pp;
            out[(v * NTIME + t) * 2 + 1] = ps;
            s_cpos[wave][(head + 25) & 31] = pp;
            s_cspd[wave][(head + 25) & 31] = ps;
        }
        head = (head + 1) & 31;
        __syncthreads();
    }
}

extern "C" void kernel_launch(void* const* d_in, const int* in_sizes, int n_in,
                              void* d_out, int out_size, void* d_ws, size_t ws_size,
                              hipStream_t stream) {
    const float* lead = (const float*)d_in[0];
    const float* cur  = (const float*)d_in[1];
    // d_in[2] = mask : unused by the reference computation
    const float* w1   = (const float*)d_in[3];
    const float* b1   = (const float*)d_in[4];
    const float* w2   = (const float*)d_in[5];
    const float* b2   = (const float*)d_in[6];
    const float* d2w  = (const float*)d_in[7];
    const float* d2b  = (const float*)d_in[8];
    const float* d1w  = (const float*)d_in[9];
    const float* d1b  = (const float*)d_in[10];
    // d_in[11] = nt (== 256, compile-time constant here)
    float* out = (float*)d_out;

    dim3 grid(NVEH / WAVES_PER_BLOCK);
    dim3 block(32 * WAVES_PER_BLOCK);
    rnncf_kernel<<<grid, block, 0, stream>>>(lead, cur, w1, b1, w2, b2,
                                             d2w, d2b, d1w, d1b, out);
}
